// GroupedQueryAttention_12463995093508
// MI455X (gfx1250) — matrix operand order repaired and checked by reading, not compiled, not run on hardware
//
#include <hip/hip_runtime.h>
#include <math.h>
#include <stdint.h>

// ---------------- types ----------------
typedef __bf16 bf16_t;
typedef __attribute__((ext_vector_type(16))) __bf16 v16bf;
typedef __attribute__((ext_vector_type(8)))  __bf16 v8bf;
typedef __attribute__((ext_vector_type(4)))  __bf16 v4bf;
typedef __attribute__((ext_vector_type(8)))  float  v8f;

// exact pointee type expected by the async-LDS builtin (from hipcc diagnostic):
// 'int __attribute__((vector_size(16)))' in AS1 / AS3
typedef int v4i_gcc __attribute__((vector_size(16)));
typedef __attribute__((address_space(1))) v4i_gcc gas_v4i;
typedef __attribute__((address_space(3))) v4i_gcc las_v4i;

#define N_HEAD   32
#define N_KV_HEAD 8
#define HEAD_DIM 64
#define B_SZ     2
#define T_SZ     2048
#define C_SZ     2048
#define KV_DIM   (N_KV_HEAD * HEAD_DIM)   // 512

#define HAS_ASYNC_LDS __has_builtin(__builtin_amdgcn_global_load_async_to_lds_b128)

// 16-byte global->LDS copy; async path uses CDNA5 GLOBAL_LOAD_ASYNC_TO_LDS_B128.
static __device__ __forceinline__ void cp16_g2l(const bf16_t* g, bf16_t* l) {
#if HAS_ASYNC_LDS
    __builtin_amdgcn_global_load_async_to_lds_b128(
        (gas_v4i*)(uintptr_t)g,
        (las_v4i*)(uint32_t)(uintptr_t)l,
        0, 0);
#else
    *(v8bf*)l = *(const v8bf*)g;
#endif
}
static __device__ __forceinline__ void wait_async_stage() {
#if HAS_ASYNC_LDS
 #if __has_builtin(__builtin_amdgcn_s_wait_asynccnt)
    __builtin_amdgcn_s_wait_asynccnt(0);
 #else
    asm volatile("s_wait_asynccnt 0x0" ::: "memory");
 #endif
#endif
}

// ---- WMMA fragment loaders (contiguous-k order shared by A and B, wave32) ----
// A 16x32: lane = row, K contiguous [lhi*16, lhi*16+16)
static __device__ __forceinline__ v16bf load_afrag(const bf16_t* p, int lhi) {
    union { v16bf v; v8bf h[2]; } u;
    u.h[0] = *(const v8bf*)(p + lhi * 16);
    u.h[1] = *(const v8bf*)(p + lhi * 16 + 8);
    return u.v;
}
// B 32x16: lane = column, K contiguous [lhi*16, lhi*16+16)
static __device__ __forceinline__ v16bf load_bfrag(const bf16_t* p, int lhi) {
    return *(const v16bf*)(p + lhi * 16);
}

// ================= prep: f32 -> bf16 (vectorized) =================
__global__ void cvt_f32_bf16(const float* __restrict__ in, bf16_t* __restrict__ out, int n4)
{
    int i = blockIdx.x * blockDim.x + threadIdx.x;
    if (i >= n4) return;
    float4 v = ((const float4*)in)[i];
    v4bf o = { (bf16_t)v.x, (bf16_t)v.y, (bf16_t)v.z, (bf16_t)v.w };
    ((v4bf*)out)[i] = o;
}

// ================= prep: transpose + convert: out[n*K+k] = in[k*N+n] =================
__global__ __launch_bounds__(256)
void transpose_cvt(const float* __restrict__ in, bf16_t* __restrict__ out, int K, int N)
{
    __shared__ float tile[32][33];
    int k0 = blockIdx.x * 32, n0 = blockIdx.y * 32;
    int tx = threadIdx.x & 31, ty = threadIdx.x >> 5;   // 32 x 8
    #pragma unroll
    for (int i = 0; i < 32; i += 8)
        tile[ty + i][tx] = in[(size_t)(k0 + ty + i) * N + n0 + tx];
    __syncthreads();
    #pragma unroll
    for (int i = 0; i < 32; i += 8)
        out[(size_t)(n0 + ty + i) * K + k0 + tx] = (bf16_t)tile[tx][ty + i];
}

// ================= WMMA GEMM: C[M,N] = A[M,K] * Bt[N,K]^T =================
// A, Bt bf16 row-major. 256 threads = 8 waves; block tile 128x128, K-step 32.
// Wave (mstrip, nhalf): 32 rows x 64 cols -> 2 A-frags x 4 B-frags = 8 WMMA/step.
template <typename OUT_T>
__global__ __launch_bounds__(256)
void gemm_bf16(const bf16_t* __restrict__ A, const bf16_t* __restrict__ Bt,
               OUT_T* __restrict__ C, int M, int N, int K)
{
    __shared__ bf16_t Xs[128 * 32];   // [row][k]
    __shared__ bf16_t Ws[128 * 32];   // [n][k]
    const int tid  = threadIdx.x;
    const int wave = tid >> 5;
    const int lane = tid & 31;
    const int lrow = lane & 15, lhi = lane >> 4;
    const int mstrip = (wave & 3) * 32;
    const int nhalf  = (wave >> 2) * 64;
    const int m0 = blockIdx.x * 128;
    const int n0 = blockIdx.y * 128;

    v8f acc[2][4];
    #pragma unroll
    for (int ms = 0; ms < 2; ++ms)
        #pragma unroll
        for (int nc = 0; nc < 4; ++nc)
            #pragma unroll
            for (int r = 0; r < 8; ++r) acc[ms][nc][r] = 0.0f;

    for (int k0 = 0; k0 < K; k0 += 32) {
        // stage both tiles: 512 x 16B chunks each; every thread copies 2+2
        #pragma unroll
        for (int i = 0; i < 2; ++i) {
            int c = tid * 2 + i;              // 0..511
            int r = c >> 2, q = (c & 3) * 8;  // row, element offset (8 bf16 = 16B)
            cp16_g2l(A  + (size_t)(m0 + r) * K + k0 + q, &Xs[r * 32 + q]);
            cp16_g2l(Bt + (size_t)(n0 + r) * K + k0 + q, &Ws[r * 32 + q]);
        }
        wait_async_stage();
        __syncthreads();

        v16bf a0 = load_afrag(&Xs[(mstrip + lrow) * 32], lhi);
        v16bf a1 = load_afrag(&Xs[(mstrip + 16 + lrow) * 32], lhi);
        #pragma unroll
        for (int nc = 0; nc < 4; ++nc) {
            v16bf b = load_bfrag(&Ws[(nhalf + nc * 16 + lrow) * 32], lhi);
            acc[0][nc] = __builtin_amdgcn_wmma_f32_16x16x32_bf16(
                false, a0, false, b, (short)0, acc[0][nc], false, false);
            acc[1][nc] = __builtin_amdgcn_wmma_f32_16x16x32_bf16(
                false, a1, false, b, (short)0, acc[1][nc], false, false);
        }
        __syncthreads();
    }
    // epilogue (D layout: vgpr r -> row r + 8*lhi, col = lrow)
    #pragma unroll
    for (int ms = 0; ms < 2; ++ms)
        #pragma unroll
        for (int nc = 0; nc < 4; ++nc)
            #pragma unroll
            for (int r = 0; r < 8; ++r) {
                size_t row = (size_t)(m0 + mstrip + ms * 16 + r + 8 * lhi);
                size_t col = (size_t)(n0 + nhalf + nc * 16 + lrow);
                C[row * N + col] = (OUT_T)acc[ms][nc][r];
            }
}

// ================= RoPE (in-place on bf16 [B*T, nheads*64]) =================
__global__ void rope_kernel(bf16_t* __restrict__ Q, int nheads, int total)
{
    int idx = blockIdx.x * blockDim.x + threadIdx.x;
    if (idx >= total) return;
    int pair = idx & 31;
    int h    = (idx >> 5) % nheads;
    int row  = idx / (32 * nheads);
    int t    = row & (T_SZ - 1);
    float inv = powf(10000.0f, -(float)pair / 32.0f);  // base^(-2i/64)
    float ang = (float)t * inv;
    float c = cosf(ang), s = sinf(ang);
    size_t base = (size_t)row * (nheads * HEAD_DIM) + h * HEAD_DIM + pair;
    float x1 = (float)Q[base], x2 = (float)Q[base + 32];
    Q[base]      = (bf16_t)(x1 * c - x2 * s);
    Q[base + 32] = (bf16_t)(x1 * s + x2 * c);
}

// ================= Flash attention (causal, GQA n_rep=4) =================
// 1 wave per (b, head, 16-row query tile). 32-key steps; WMMA bf16; bf16 out.
__global__ __launch_bounds__(32)
void flash_attn(const bf16_t* __restrict__ Qb, const bf16_t* __restrict__ Kb,
                const bf16_t* __restrict__ Vb, bf16_t* __restrict__ O)
{
    __shared__ bf16_t Pt[16 * 32];    // P tile   [qrow][key]
    __shared__ bf16_t Vt[64 * 32];    // V^T tile [dim][key]
    const int lane = threadIdx.x;
    const int lrow = lane & 15, lhi = lane >> 4;
    const int m0  = blockIdx.x * 16;
    const int h   = blockIdx.y;
    const int b   = blockIdx.z;
    const int kvh = h >> 2;                        // h // n_rep
    const float scale = 0.125f;                    // 1/sqrt(64)

    const bf16_t* qrow = Qb + (size_t)(b * T_SZ + m0 + lrow) * C_SZ + h * HEAD_DIM;
    const v16bf aq0 = load_afrag(qrow, lhi);        // dims 0..31
    const v16bf aq1 = load_afrag(qrow + 32, lhi);   // dims 32..63

    float rM[8], rL[8];
    v8f acc[4];
    #pragma unroll
    for (int r = 0; r < 8; ++r) { rM[r] = -1e30f; rL[r] = 0.0f; }
    #pragma unroll
    for (int nc = 0; nc < 4; ++nc)
        #pragma unroll
        for (int r = 0; r < 8; ++r) acc[nc][r] = 0.0f;

    for (int j0 = 0; j0 <= m0 + 15; j0 += 32) {
        // ---- stage V^T: 32 keys x 64 dims -> Vt[dim][key]
        {
            const bf16_t* vrow = Vb + (size_t)(b * T_SZ + j0 + lane) * KV_DIM + kvh * HEAD_DIM;
            #pragma unroll
            for (int d8 = 0; d8 < 8; ++d8) {
                v8bf vv = *(const v8bf*)(vrow + d8 * 8);
                #pragma unroll
                for (int j = 0; j < 8; ++j)
                    Vt[(d8 * 8 + j) * 32 + lane] = vv[j];
            }
        }
        // ---- S = Q K^T  (two 16x16 subtiles, each 2 WMMAs over head dim)
        v8f S[2];
        #pragma unroll
        for (int s = 0; s < 2; ++s) {
            const bf16_t* krow =
                Kb + (size_t)(b * T_SZ + j0 + s * 16 + lrow) * KV_DIM + kvh * HEAD_DIM;
            v16bf bk0 = load_bfrag(krow, lhi);
            v16bf bk1 = load_bfrag(krow + 32, lhi);
            v8f z;
            #pragma unroll
            for (int r = 0; r < 8; ++r) z[r] = 0.0f;
            z = __builtin_amdgcn_wmma_f32_16x16x32_bf16(false, aq0, false, bk0, (short)0, z, false, false);
            z = __builtin_amdgcn_wmma_f32_16x16x32_bf16(false, aq1, false, bk1, (short)0, z, false, false);
            S[s] = z;
        }
        // ---- mask + online softmax (per-row stats via 16-lane xor reductions)
        float alpha[8], p0[8], p1[8];
        #pragma unroll
        for (int r = 0; r < 8; ++r) {
            int m = m0 + r + 8 * lhi;
            float v0 = (j0 + lrow      <= m) ? S[0][r] * scale : -1e30f;
            float v1 = (j0 + 16 + lrow <= m) ? S[1][r] * scale : -1e30f;
            float vmax = fmaxf(v0, v1);
            #pragma unroll
            for (int msk = 1; msk < 16; msk <<= 1)
                vmax = fmaxf(vmax, __shfl_xor(vmax, msk, 32));
            float Mn = fmaxf(rM[r], vmax);
            alpha[r] = __expf(rM[r] - Mn);
            rM[r] = Mn;
            p0[r] = __expf(v0 - Mn);
            p1[r] = __expf(v1 - Mn);
            float ps = p0[r] + p1[r];
            #pragma unroll
            for (int msk = 1; msk < 16; msk <<= 1)
                ps += __shfl_xor(ps, msk, 32);
            rL[r] = rL[r] * alpha[r] + ps;
        }
        #pragma unroll
        for (int nc = 0; nc < 4; ++nc)
            #pragma unroll
            for (int r = 0; r < 8; ++r) acc[nc][r] *= alpha[r];
        // ---- P to LDS (D layout -> row-major [16][32])
        __syncthreads();
        #pragma unroll
        for (int r = 0; r < 8; ++r) {
            Pt[(r + 8 * lhi) * 32 + lrow]      = (bf16_t)p0[r];
            Pt[(r + 8 * lhi) * 32 + 16 + lrow] = (bf16_t)p1[r];
        }
        __syncthreads();
        // ---- O += P @ V   (4 WMMAs across the 64 output dims)
        v16bf ap = load_afrag(&Pt[lrow * 32], lhi);
        #pragma unroll
        for (int nc = 0; nc < 4; ++nc) {
            v16bf bv = load_bfrag(&Vt[(nc * 16 + lrow) * 32], lhi);
            acc[nc] = __builtin_amdgcn_wmma_f32_16x16x32_bf16(
                false, ap, false, bv, (short)0, acc[nc], false, false);
        }
        __syncthreads();
    }
    // ---- finalize: O / l, store bf16 [B*T, C]
    #pragma unroll
    for (int nc = 0; nc < 4; ++nc)
        #pragma unroll
        for (int r = 0; r < 8; ++r) {
            int m = m0 + r + 8 * lhi;
            O[(size_t)(b * T_SZ + m) * C_SZ + h * HEAD_DIM + nc * 16 + lrow] =
                (bf16_t)(acc[nc][r] / rL[r]);
        }
}

// ================= launcher =================
extern "C" void kernel_launch(void* const* d_in, const int* in_sizes, int n_in,
                              void* d_out, int out_size, void* d_ws, size_t ws_size,
                              hipStream_t stream)
{
    (void)in_sizes; (void)n_in; (void)out_size; (void)ws_size;
    const float* x  = (const float*)d_in[0];
    const float* Wq = (const float*)d_in[1];
    const float* Wk = (const float*)d_in[2];
    const float* Wv = (const float*)d_in[3];
    const float* Wo = (const float*)d_in[4];
    float* out = (float*)d_out;

    const int M = B_SZ * T_SZ;  // 4096
    bf16_t* xb  = (bf16_t*)d_ws;                       // [M, 2048]
    bf16_t* Wqt = xb  + (size_t)M * C_SZ;              // [2048, 2048] (W^T)
    bf16_t* Wkt = Wqt + (size_t)C_SZ * C_SZ;           // [512, 2048]
    bf16_t* Wvt = Wkt + (size_t)KV_DIM * C_SZ;         // [512, 2048]
    bf16_t* Wot = Wvt + (size_t)KV_DIM * C_SZ;         // [2048, 2048]
    bf16_t* Qb  = Wot + (size_t)C_SZ * C_SZ;           // [M, 2048]
    bf16_t* Kb  = Qb  + (size_t)M * C_SZ;              // [M, 512]
    bf16_t* Vb  = Kb  + (size_t)M * KV_DIM;            // [M, 512]
    bf16_t* Ob  = Vb  + (size_t)M * KV_DIM;            // [M, 2048]

    // prep: convert x, transpose+convert weights
    {
        int n4 = (M * C_SZ) / 4;
        cvt_f32_bf16<<<(n4 + 255) / 256, 256, 0, stream>>>(x, xb, n4);
    }
    transpose_cvt<<<dim3(C_SZ / 32, C_SZ / 32),   256, 0, stream>>>(Wq, Wqt, C_SZ, C_SZ);
    transpose_cvt<<<dim3(C_SZ / 32, KV_DIM / 32), 256, 0, stream>>>(Wk, Wkt, C_SZ, KV_DIM);
    transpose_cvt<<<dim3(C_SZ / 32, KV_DIM / 32), 256, 0, stream>>>(Wv, Wvt, C_SZ, KV_DIM);
    transpose_cvt<<<dim3(C_SZ / 32, C_SZ / 32),   256, 0, stream>>>(Wo, Wot, C_SZ, C_SZ);

    // projections
    gemm_bf16<bf16_t><<<dim3(M / 128, C_SZ / 128),   256, 0, stream>>>(xb, Wqt, Qb, M, C_SZ,   C_SZ);
    gemm_bf16<bf16_t><<<dim3(M / 128, KV_DIM / 128), 256, 0, stream>>>(xb, Wkt, Kb, M, KV_DIM, C_SZ);
    gemm_bf16<bf16_t><<<dim3(M / 128, KV_DIM / 128), 256, 0, stream>>>(xb, Wvt, Vb, M, KV_DIM, C_SZ);

    // RoPE
    int totq = M * N_HEAD * 32;
    rope_kernel<<<(totq + 255) / 256, 256, 0, stream>>>(Qb, N_HEAD, totq);
    int totk = M * N_KV_HEAD * 32;
    rope_kernel<<<(totk + 255) / 256, 256, 0, stream>>>(Kb, N_KV_HEAD, totk);

    // attention
    flash_attn<<<dim3(T_SZ / 16, N_HEAD, B_SZ), 32, 0, stream>>>(Qb, Kb, Vb, Ob);

    // output projection (f32 out)
    gemm_bf16<float><<<dim3(M / 128, C_SZ / 128), 256, 0, stream>>>(Ob, Wot, out, M, C_SZ, C_SZ);
}
